// Net_MP_11390253269739
// MI455X (gfx1250) — matrix operand order repaired and checked by reading, not compiled, not run on hardware
//
#include <hip/hip_runtime.h>
#include <hip/hip_bf16.h>

typedef __attribute__((ext_vector_type(16))) _Float16 v16h;
typedef __attribute__((ext_vector_type(8)))  float    v8f;

#define F_HID 32

// ---------------------------------------------------------------------------
// Prep: rearrange nn_w2 [cin*32,16] (+ bias rows from b2) into WMMA B-fragment
// order.  Logical B[kappa][o], kappa = i*16+k (U part, kappa<KU=cin*16), then
// bias rows kappa-KU = i (value b2[i*32+o]), zero-padded to 32*NCHUNK.
// Fragment layout (wave32 f16 B, 32xK x 16): lane = N%16 (+16 selects K-half),
// element t <-> K = 32*c + 16*(t/8) + 8*(lane/16) + t%8.
// flat index = ((c*2+nt)*32 + lane)*16 + t
// ---------------------------------------------------------------------------
__global__ void prep_bfrag(const float* __restrict__ w2, const float* __restrict__ b2,
                           _Float16* __restrict__ bf, int KU, int CIN, int NCHUNK) {
  int idx = blockIdx.x * blockDim.x + threadIdx.x;
  int total = NCHUNK * 2 * 32 * 16;
  if (idx >= total) return;
  int t    = idx & 15;
  int lane = (idx >> 4) & 31;
  int nt   = (idx >> 9) & 1;
  int c    = idx >> 10;
  int hf   = lane >> 4;
  int N    = nt * 16 + (lane & 15);
  int K    = 32 * c + ((t < 8) ? 0 : 16) + 8 * hf + (t & 7);
  float val = 0.f;
  if (K < KU) {
    int i = K >> 4, k = K & 15;
    val = w2[(i * F_HID + N) * 16 + k];
  } else if (K < KU + CIN) {
    val = b2[(K - KU) * F_HID + N];
  }
  bf[idx] = (_Float16)val;
}

// ---------------------------------------------------------------------------
// Edge message kernel (NNConv message + scatter).  One wave per 16 edges.
// A tile [16 edges x 32 K] built in registers per chunk from v (x[src]) and
// h = relu(edge_attr @ w1^T + b1); two WMMAs per chunk (N tiles 0-15 / 16-31).
// Accumulate with global f32 atomics into num[dst,32] (+ cnt on layer 1).
// ---------------------------------------------------------------------------
template <int CIN, int NCHUNK>
__global__ __launch_bounds__(128) void edge_msg_kernel(
    const float* __restrict__ xin,        // [N, CIN]
    const int*   __restrict__ eidx,       // [2, E]
    const float* __restrict__ eattr,      // [E, 3]
    const float* __restrict__ w1,         // [16, 3]
    const float* __restrict__ b1,         // [16]
    const _Float16* __restrict__ bfragG,  // [NCHUNK*2*32] x v16h
    float* __restrict__ num,              // [N, 32]
    float* __restrict__ cnt,              // [N]
    int E, int addCnt)
{
  constexpr int KU = CIN * 16;
  __shared__ v16h sB[NCHUNK * 2 * 32];

  { // cooperative load of pre-swizzled B fragments into LDS
    const uint4* g = (const uint4*)bfragG;
    uint4* s = (uint4*)sB;
    const int nv = NCHUNK * 2 * 32 * 2;   // 16B units
    for (int i = threadIdx.x; i < nv; i += blockDim.x) s[i] = g[i];
  }
  __syncthreads();

  const int wv   = threadIdx.x >> 5;
  const int lane = threadIdx.x & 31;
  const int hf   = lane >> 4;     // which half-wave
  const int m    = lane & 15;     // edge slot within tile
  const long base = ((long)blockIdx.x * (blockDim.x >> 5) + wv) * 16;
  if (base >= E) return;          // wave-uniform guard (EXEC stays all-ones)

  const int e = (int)base + m;
  const bool valid = e < E;

  // --- edge attr + hidden MLP: h[k] = relu(ea . w1[k] + b1[k]); keep the
  //     8-entry half this lane group needs (hh[j] == h[8*hf + j]).
  float ea0 = 0.f, ea1 = 0.f, ea2 = 0.f;
  int src = 0;
  if (valid) {
    src = eidx[e];
    ea0 = eattr[e * 3 + 0];
    ea1 = eattr[e * 3 + 1];
    ea2 = eattr[e * 3 + 2];
  }
  _Float16 hh[8];
#pragma unroll
  for (int j = 0; j < 8; ++j) {
    const int k0 = j, k1 = j + 8;
    float f0 = ea0 * w1[k0 * 3 + 0] + ea1 * w1[k0 * 3 + 1] + ea2 * w1[k0 * 3 + 2] + b1[k0];
    float f1 = ea0 * w1[k1 * 3 + 0] + ea1 * w1[k1 * 3 + 1] + ea2 * w1[k1 * 3 + 2] + b1[k1];
    f0 = f0 > 0.f ? f0 : 0.f;
    f1 = f1 > 0.f ? f1 : 0.f;
    hh[j] = (_Float16)(hf ? f1 : f0);
  }

  // --- source-node features v (shared by lanes L and L+16)
  float vf[CIN];
#pragma unroll
  for (int i = 0; i < CIN; ++i) vf[i] = 0.f;
  if (valid) {
#pragma unroll
    for (int i = 0; i < CIN; ++i) vf[i] = xin[(long)src * CIN + i];
  }
  _Float16 vh[CIN];
#pragma unroll
  for (int i = 0; i < CIN; ++i) vh[i] = (_Float16)vf[i];

  // --- GEMM over K chunks of 32: C[16x32] += A[16x32] * B[32x32]
  v8f acc0 = {};
  v8f acc1 = {};
#pragma unroll
  for (int c = 0; c < NCHUNK; ++c) {
    v16h a;
#pragma unroll
    for (int t = 0; t < 16; ++t) {
      const int s  = t & 7;
      const int kb = 32 * c + ((t < 8) ? 0 : 16);  // kappa = kb + 8*hf + s
      _Float16 av;
      if (kb + s < KU) {
        // U part: i = kb>>4 (uniform across halves), k-half folded into hh
        const int ib = kb >> 4;
        av = vh[(ib < CIN) ? ib : 0] * hh[s];
      } else {
        // bias rows (a = v[kappa-KU]) or zero padding
        const int j = kb + s - KU;               // index for hf==0
        const _Float16 lo = (j     < CIN) ? vh[(j     < CIN) ? j     : 0] : (_Float16)0.f;
        const _Float16 hi = (j + 8 < CIN) ? vh[(j + 8 < CIN) ? j + 8 : 0] : (_Float16)0.f;
        av = hf ? hi : lo;
      }
      a[t] = av;
    }
    const v16h b0 = sB[(c * 2 + 0) * 32 + lane];
    const v16h b1v = sB[(c * 2 + 1) * 32 + lane];
    acc0 = __builtin_amdgcn_wmma_f32_16x16x32_f16(false, a, false, b0,
                                                  (short)0, acc0, false, false);
    acc1 = __builtin_amdgcn_wmma_f32_16x16x32_f16(false, a, false, b1v,
                                                  (short)0, acc1, false, false);
  }

  // --- scatter: D layout row M = r + 8*hf, col N = lane&15 (+16 for acc1)
  const int* dstp = eidx + E;
  const int col = lane & 15;
#pragma unroll
  for (int r = 0; r < 8; ++r) {
    const long er = base + 8 * hf + r;
    if (er < E) {
      const int d = dstp[er];
      atomicAdd(&num[(long)d * F_HID + col],      acc0[r]);
      atomicAdd(&num[(long)d * F_HID + 16 + col], acc1[r]);
    }
  }
  if (addCnt && valid && lane < 16) atomicAdd(&cnt[dstp[e]], 1.0f);
}

// ---------------------------------------------------------------------------
// Node update: x_out = relu(num/max(cnt,1) + x_in @ root + bias)
// ---------------------------------------------------------------------------
__global__ void node_update(const float* __restrict__ num, const float* __restrict__ cnt,
                            const float* __restrict__ xin, const float* __restrict__ root,
                            const float* __restrict__ bias, float* __restrict__ xout,
                            int n, int cin) {
  long i = (long)blockIdx.x * blockDim.x + threadIdx.x;
  if (i >= (long)n * F_HID) return;
  const int node = (int)(i >> 5);
  const int o = (int)(i & 31);
  float c = cnt[node];
  c = c > 1.f ? c : 1.f;
  float s = num[i] / c + bias[o];
  for (int k = 0; k < cin; ++k) s += xin[(long)node * cin + k] * root[k * F_HID + o];
  xout[i] = s > 0.f ? s : 0.f;
}

// ---------------------------------------------------------------------------
// Head: out = relu(x2[mask] @ fc1^T + b1) @ fc2^T + b2
// ---------------------------------------------------------------------------
__global__ void head_kernel(const float* __restrict__ x2, const int* __restrict__ mask,
                            const float* __restrict__ fc1w, const float* __restrict__ fc1b,
                            const float* __restrict__ fc2w, const float* __restrict__ fc2b,
                            float* __restrict__ out, int M) {
  int mi = blockIdx.x * blockDim.x + threadIdx.x;
  if (mi >= M) return;
  const int node = mask[mi];
  float xm[32];
#pragma unroll
  for (int j = 0; j < 32; ++j) xm[j] = x2[(long)node * F_HID + j];
  float o0 = fc2b[0], o1 = fc2b[1];
  for (int o = 0; o < 32; ++o) {
    float h = fc1b[o];
#pragma unroll
    for (int j = 0; j < 32; ++j) h += xm[j] * fc1w[o * 32 + j];
    h = h > 0.f ? h : 0.f;
    o0 += h * fc2w[o];
    o1 += h * fc2w[32 + o];
  }
  out[mi * 2 + 0] = o0;
  out[mi * 2 + 1] = o1;
}

// ---------------------------------------------------------------------------
extern "C" void kernel_launch(void* const* d_in, const int* in_sizes, int n_in,
                              void* d_out, int out_size, void* d_ws, size_t ws_size,
                              hipStream_t stream) {
  const float* x       = (const float*)d_in[0];
  const int*   eidx    = (const int*)  d_in[1];
  const float* eattr   = (const float*)d_in[2];
  const int*   mask    = (const int*)  d_in[3];
  const float* nn1_w1  = (const float*)d_in[4];
  const float* nn1_b1  = (const float*)d_in[5];
  const float* nn1_w2  = (const float*)d_in[6];
  const float* nn1_b2  = (const float*)d_in[7];
  const float* c1_root = (const float*)d_in[8];
  const float* c1_bias = (const float*)d_in[9];
  const float* nn3_w1  = (const float*)d_in[10];
  const float* nn3_b1  = (const float*)d_in[11];
  const float* nn3_w2  = (const float*)d_in[12];
  const float* nn3_b2  = (const float*)d_in[13];
  const float* c3_root = (const float*)d_in[14];
  const float* c3_bias = (const float*)d_in[15];
  const float* fc1_w   = (const float*)d_in[16];
  const float* fc1_b   = (const float*)d_in[17];
  const float* fc2_w   = (const float*)d_in[18];
  const float* fc2_b   = (const float*)d_in[19];

  const int N = in_sizes[0] / 5;
  const int E = in_sizes[2] / 3;
  const int M = in_sizes[3];

  // workspace carve-up
  char* ws = (char*)d_ws;
  size_t off = 0;
  auto carve = [&](size_t bytes) -> void* {
    void* p = ws + off;
    off += (bytes + 255) & ~(size_t)255;
    return p;
  };
  float*    x1     = (float*)   carve((size_t)N * F_HID * sizeof(float));
  float*    x2     = (float*)   carve((size_t)N * F_HID * sizeof(float));
  float*    num    = (float*)   carve((size_t)N * F_HID * sizeof(float));
  float*    cnt    = (float*)   carve((size_t)N * sizeof(float));
  _Float16* bfrag1 = (_Float16*)carve((size_t)3  * 2 * 32 * 16 * sizeof(_Float16));
  _Float16* bfrag3 = (_Float16*)carve((size_t)17 * 2 * 32 * 16 * sizeof(_Float16));

  // pre-swizzle B matrices into WMMA fragment order
  {
    int t1 = 3 * 2 * 32 * 16;
    int t3 = 17 * 2 * 32 * 16;
    prep_bfrag<<<(t1 + 255) / 256, 256, 0, stream>>>(nn1_w2, nn1_b2, bfrag1, 80, 5, 3);
    prep_bfrag<<<(t3 + 255) / 256, 256, 0, stream>>>(nn3_w2, nn3_b2, bfrag3, 512, 32, 17);
  }

  hipMemsetAsync(num, 0, (size_t)N * F_HID * sizeof(float), stream);
  hipMemsetAsync(cnt, 0, (size_t)N * sizeof(float), stream);

  const int egrid = (E + 63) / 64;  // 4 waves/block * 16 edges/wave

  // layer 1:  K = 80 (U) + 5 (bias) -> 3 chunks of 32
  edge_msg_kernel<5, 3><<<egrid, 128, 0, stream>>>(x, eidx, eattr, nn1_w1, nn1_b1,
                                                   bfrag1, num, cnt, E, 1);
  node_update<<<((long)N * F_HID + 255) / 256, 256, 0, stream>>>(num, cnt, x, c1_root,
                                                                 c1_bias, x1, N, 5);

  hipMemsetAsync(num, 0, (size_t)N * F_HID * sizeof(float), stream);

  // layer 2:  K = 512 (U) + 32 (bias) -> 17 chunks of 32
  edge_msg_kernel<32, 17><<<egrid, 128, 0, stream>>>(x1, eidx, eattr, nn3_w1, nn3_b1,
                                                     bfrag3, num, cnt, E, 0);
  node_update<<<((long)N * F_HID + 255) / 256, 256, 0, stream>>>(num, cnt, x1, c3_root,
                                                                 c3_bias, x2, N, 32);

  // head
  head_kernel<<<(M + 255) / 256, 256, 0, stream>>>(x2, mask, fc1_w, fc1_b, fc2_w, fc2_b,
                                                   (float*)d_out, M);
}
